// LiGRU_40982577938581
// MI455X (gfx1250) — matrix operand order repaired and checked by reading, not compiled, not run on hardware
//
#include <hip/hip_runtime.h>
#include <stdint.h>

typedef __bf16 bf16;
typedef __attribute__((ext_vector_type(16))) __bf16 v16bf;
typedef __attribute__((ext_vector_type(8)))  __bf16 v8bf;
typedef __attribute__((ext_vector_type(8)))  float  v8f;
typedef __attribute__((ext_vector_type(4)))  unsigned int u32x4;
typedef __attribute__((ext_vector_type(8)))  int i32x8;
typedef __attribute__((ext_vector_type(4)))  int i32x4;

#define B_   16
#define T_   1000
#define H_   512
#define G2H_ 1024
#define L_   4
#define M_   (B_*T_)          /* 16000 rows, row = b*T + t */
#define EPSL 1e-5f
#define NWG  8                /* workgroups in the recurrent kernel */

#define AS1 __attribute__((address_space(1)))
#define AS3 __attribute__((address_space(3)))

/* ---- CDNA5 feature detection (device pass only) + compile-time probes ---- */
#if defined(__HIP_DEVICE_COMPILE__)
  #if __has_builtin(__builtin_amdgcn_tensor_load_to_lds) && __has_builtin(__builtin_amdgcn_s_wait_tensorcnt)
    #define HAVE_TDM 1
  #else
    #define HAVE_TDM 0
    #warning "probe: TDM tensor_load_to_lds / s_wait_tensorcnt builtins MISSING - using plain-load fallback"
  #endif
  #if __has_builtin(__builtin_amdgcn_global_load_async_to_lds_b128) && __has_builtin(__builtin_amdgcn_s_wait_asynccnt)
    #define HAVE_ASYNC 1
  #else
    #define HAVE_ASYNC 0
    #warning "probe: global_load_async_to_lds_b128 / s_wait_asynccnt builtins MISSING - using plain-load fallback"
  #endif
  #if __has_builtin(__builtin_amdgcn_s_cluster_barrier)
    #define CLUSTER_BARRIER() __builtin_amdgcn_s_cluster_barrier()
  #else
    #define CLUSTER_BARRIER()
    #warning "probe: s_cluster_barrier builtin MISSING"
  #endif
  #define DEV_SLEEP() __builtin_amdgcn_s_sleep(1)
#else
  #define HAVE_TDM 0
  #define HAVE_ASYNC 0
  #define CLUSTER_BARRIER()
  #define DEV_SLEEP()
#endif

__device__ __forceinline__ v16bf combine16(v8bf lo, v8bf hi) {
  v16bf r;
#pragma unroll
  for (int i = 0; i < 8; ++i) { r[i] = lo[i]; r[8 + i] = hi[i]; }
  return r;
}

__device__ __forceinline__ v16bf loadB(const bf16* p, int stride) {
  v16bf r;
#pragma unroll
  for (int i = 0; i < 8; ++i) { r[i] = p[i * stride]; r[8 + i] = p[(16 + i) * stride]; }
  return r;
}

#if HAVE_ASYNC
/* 16B global -> LDS async copy (per-lane addresses, tracked by ASYNCcnt).
   Builtin prototype is (V4i AS1*, V4i AS3*, imm offset, imm cpol).
   Generic->AS casts via integer round-trip: LDS offset = low 32 bits of the
   generic shared-aperture address (ISA 10.2 aperture rules). */
__device__ __forceinline__ void async_cp16(void* lds, const void* g) {
  __builtin_amdgcn_global_load_async_to_lds_b128(
      (AS1 i32x4*)(uintptr_t)g, (AS3 i32x4*)(uint32_t)(uintptr_t)lds, 0, 0);
}
#endif

/* device-wide split barrier: cluster-barrier fast path (NOP when not
   cluster-dispatched) + deterministic release/acquire counter sync. */
__device__ __forceinline__ void wg_grid_sync(unsigned* cnt, unsigned target) {
  __syncthreads();
  if (threadIdx.x == 0) {
    CLUSTER_BARRIER();  /* s_barrier_signal -3 ; s_barrier_wait -3 */
    __hip_atomic_fetch_add(cnt, 1u, __ATOMIC_RELEASE, __HIP_MEMORY_SCOPE_AGENT);
    while (__hip_atomic_load(cnt, __ATOMIC_ACQUIRE, __HIP_MEMORY_SCOPE_AGENT) < target) {
      DEV_SLEEP();
    }
  }
  __syncthreads();
  __threadfence();
}

/* ---------------- prep kernels ---------------- */
__global__ void k_f32_to_bf16(const float* __restrict__ src, bf16* __restrict__ dst, int n) {
  int i = blockIdx.x * 256 + threadIdx.x;
  if (i < n) dst[i] = (bf16)src[i];
}

/* src: [L][2H][H] row-major -> dst: [L][H][2H] (K-major, bf16) */
__global__ void k_transpose_to_bf16(const float* __restrict__ src, bf16* __restrict__ dst) {
  int i = blockIdx.x * 256 + threadIdx.x;
  if (i < L_ * G2H_ * H_) {
    int l = i / (G2H_ * H_);
    int r = i - l * (G2H_ * H_);
    int n = r / H_;
    int k = r - n * H_;
    dst[(size_t)l * H_ * G2H_ + (size_t)k * G2H_ + n] = (bf16)src[i];
  }
}

/* ---------------- feedforward GEMM: Wout[M,2H] = A[M,H] @ BT[H,2H] ----------------
   grid (125, 8): 8 waves/WG each own one 16-row M tile and the block's 128-wide
   N strip. Fragment loads are software-pipelined one tile ahead of the WMMA
   consuming them so s_wait_loadcnt overlaps with matrix math. */
__global__ __launch_bounds__(256) void k_gemm_ff(
    const bf16* __restrict__ A, const bf16* __restrict__ BT, float* __restrict__ Wout) {
  const int tid  = threadIdx.x;
  const int wave = tid >> 5, lane = tid & 31;
  const int m0   = (blockIdx.x * 8 + wave) * 16;
  const int n0   = blockIdx.y * 128;
  const int koff = (lane < 16) ? 0 : 8;        /* A-fragment lane-half K offset */
  const bf16* aRow  = A + (size_t)(m0 + (lane & 15)) * H_ + koff;
  const bf16* bBase = BT + (size_t)koff * G2H_ + n0 + (lane & 15);  /* B: N = lane&15, K offset koff */

  v8f zero = {};
  v8f acc[8];
#pragma unroll
  for (int i = 0; i < 8; ++i) acc[i] = zero;

  /* prologue: fragment (kb=0, tt=0) */
  v16bf a = combine16(*(const v8bf*)(aRow), *(const v8bf*)(aRow + 16));
  v16bf b = loadB(bBase, G2H_);

  for (int kb = 0; kb < 16; ++kb) {
    const bf16* aN = aRow + (kb + 1) * 32;
    v16bf aNext = (kb < 15)
        ? combine16(*(const v8bf*)(aN), *(const v8bf*)(aN + 16)) : a;
    const bf16* bRow  = bBase + (size_t)(kb * 32) * G2H_;
    const bf16* bRowN = bBase + (size_t)((kb + 1) * 32) * G2H_;
#pragma unroll
    for (int tt = 0; tt < 8; ++tt) {
      const bf16* pn = (tt < 7) ? (bRow + (tt + 1) * 16) : bRowN;
      v16bf bNext = (kb < 15 || tt < 7)
          ? loadB(pn, G2H_) : b;
      acc[tt] = __builtin_amdgcn_wmma_f32_16x16x32_bf16(
                    false, a, false, b, (short)0, acc[tt], false, false);
      b = bNext;
    }
    a = aNext;
  }
  const int rowAdd = (lane < 16) ? 0 : 8;      /* C layout: VGPR r -> M=r(+8) */
#pragma unroll
  for (int tt = 0; tt < 8; ++tt) {
    const int col = n0 + tt * 16 + (lane & 15);
#pragma unroll
    for (int r = 0; r < 8; ++r)
      Wout[(size_t)(m0 + r + rowAdd) * G2H_ + col] = acc[tt][r];
  }
}

/* ---------------- row LayerNorm (affine) over 1024 columns ---------------- */
__global__ __launch_bounds__(256) void k_ln_rows(
    float* __restrict__ w, const float* __restrict__ gamma, const float* __restrict__ beta) {
  __shared__ float rs[256], rq[256];
  const int tid = threadIdx.x;
  float* v = w + (size_t)blockIdx.x * G2H_;
  float x[4], s = 0.f, q = 0.f;
#pragma unroll
  for (int i = 0; i < 4; ++i) { x[i] = v[tid + 256 * i]; s += x[i]; q += x[i] * x[i]; }
  rs[tid] = s; rq[tid] = q;
  __syncthreads();
  for (int off = 128; off > 0; off >>= 1) {
    if (tid < off) { rs[tid] += rs[tid + off]; rq[tid] += rq[tid + off]; }
    __syncthreads();
  }
  const float mu   = rs[0] * (1.0f / 1024.0f);
  const float var  = rq[0] * (1.0f / 1024.0f) - mu * mu;
  const float rstd = rsqrtf(var + EPSL);
#pragma unroll
  for (int i = 0; i < 4; ++i) {
    const int col = tid + 256 * i;
    v[col] = (x[i] - mu) * rstd * gamma[col] + beta[col];
  }
}

/* ---------------- recurrent scan ----------------
   8 WGs (one per WGP, 169 KB LDS each): WG g owns a-cols [64g,64g+64) and the
   matching z-cols [512+64g, ...). Its Uᵀ slice (512x128 bf16 = 128 KB) lives in
   LDS for the entire scan. Per step: TDM stages h_{t-1} (16x512 bf16) into
   padded LDS rows, 8 waves each run 16 WMMAs, LN-stats exchanged device-wide. */
__global__ __launch_bounds__(256) void k_recur(
    const bf16* __restrict__ UT,   /* [512][1024] for this layer */
    const float* __restrict__ w,   /* [M_][1024], already LN-affined */
    bf16* __restrict__ hA,         /* [B][T][H] bf16: layer activations (in-place) */
    float* __restrict__ out,       /* [B][T][H] fp32 */
    float* __restrict__ partials,  /* [2][NWG][32] */
    unsigned* __restrict__ sync1, unsigned* __restrict__ sync2, int layer) {
  extern __shared__ char smem[];
  const int UB = 0;                 /* ubuf: 512 x 136 bf16 (8-elem pad)  139264 B */
  const int HB = 139264;            /* hbuf:  16 x 520 bf16 (8-elem pad)   16640 B */
  const int HS = HB + 16640;        /* hstate 16x64 f32 */
  const int HC = HS + 4096;         /* hcand  16x64 f32 */
  const int HZ = HC + 4096;         /* z      16x64 f32 */
  const int WP = HZ + 4096;         /* wave partials 8x32 f32 */
  const int ST = WP + 1024;         /* stats 32 f32 */
  bf16*  ubuf   = (bf16*)(smem + UB);
  bf16*  hbuf   = (bf16*)(smem + HB);
  float* hstate = (float*)(smem + HS);
  float* hcand  = (float*)(smem + HC);
  float* zbuf   = (float*)(smem + HZ);
  float* wpart  = (float*)(smem + WP);
  float* stats  = (float*)(smem + ST);

  const int g = blockIdx.x, tid = threadIdx.x;
  const int wave = tid >> 5, lane = tid & 31;

  /* one-time: stage this WG's Uᵀ slice (a-cols then z-cols) into LDS */
  for (int idx = tid; idx < 512 * 16; idx += 256) {
    const int k = idx >> 4;
    const int c = (idx & 15) * 8;
    const int gc = (c < 64) ? (64 * g + c) : (512 + 64 * g + (c - 64));
#if HAVE_ASYNC
    async_cp16(ubuf + k * 136 + c, UT + (size_t)k * G2H_ + gc);
#else
    *(v8bf*)(ubuf + k * 136 + c) = *(const v8bf*)(UT + (size_t)k * G2H_ + gc);
#endif
  }
  for (int idx = tid; idx < 16 * 64; idx += 256) hstate[idx] = 0.0f;
#if HAVE_ASYNC
  __builtin_amdgcn_s_wait_asynccnt(0);
#endif
  __syncthreads();

  const int isZ     = (wave >= 4);
  const int ubufCol = isZ ? (64 + 16 * (wave - 4)) : (16 * wave);
  const int colBase = isZ ? (512 + 64 * g + 16 * (wave - 4)) : (64 * g + 16 * wave);
  const int lc      = (wave & 3) * 16 + (lane & 15);   /* local h column 0..63 */
  const int myN     = lane & 15;
  const int rowAdd  = (lane < 16) ? 0 : 8;
  const int koff    = (lane < 16) ? 0 : 8;

#if HAVE_TDM
  const unsigned ldsHbuf = __builtin_amdgcn_groupstaticsize() + (unsigned)HB;
#endif
  const unsigned baseT = (unsigned)layer * (unsigned)T_ * NWG;

  for (int t = 0; t < T_; ++t) {
    v8f acc = {};
    if (t > 0) {
#if HAVE_TDM
      if (wave == 0) {
        /* D#: 2-D tile 512x16 (2-byte elems), row stride T*H, LDS pad 4 dwords
           after every 256 dwords -> 520-element padded LDS rows. */
        unsigned long long ga = (unsigned long long)(size_t)(hA + (size_t)(t - 1) * H_);
        u32x4 g0;
        g0[0] = 1u;                                            /* count=1 */
        g0[1] = ldsHbuf;                                       /* lds_addr */
        g0[2] = (unsigned)ga;
        g0[3] = (unsigned)((ga >> 32) & 0x1FFFFFFull) | (2u << 30);  /* type=2 */
        i32x8 g1;
        g1[0] = (1 << 16) | (1 << 20) | (7 << 22) | (3 << 25); /* 2B, pad_en, ivl=256dw, amt=4dw */
        g1[1] = (512 << 16);                                   /* tensor_dim0 = 512 */
        g1[2] = (16 << 16);                                    /* tensor_dim1 = 16 */
        g1[3] = (512 << 16);                                   /* tile_dim0 = 512 */
        g1[4] = 16;                                            /* tile_dim1 = 16 */
        g1[5] = T_ * H_;                                       /* dim0 stride = 512000 */
        g1[6] = 0; g1[7] = 0;
        i32x4 zz = {};
#if __clang_major__ >= 23
        i32x8 z8 = {};
        __builtin_amdgcn_tensor_load_to_lds(g0, g1, zz, zz, z8, 0);
#else
        __builtin_amdgcn_tensor_load_to_lds(g0, g1, zz, zz, 0);
#endif
        __builtin_amdgcn_s_wait_tensorcnt(0);
      }
      __syncthreads();
#else
      for (int idx = tid; idx < 16 * 64; idx += 256) {
        const int m = idx >> 6, c = (idx & 63) * 8;
#if HAVE_ASYNC
        async_cp16(hbuf + m * 520 + c, hA + ((size_t)m * T_ + (t - 1)) * H_ + c);
#else
        *(v8bf*)(hbuf + m * 520 + c) =
            *(const v8bf*)(hA + ((size_t)m * T_ + (t - 1)) * H_ + c);
#endif
      }
#if HAVE_ASYNC
      __builtin_amdgcn_s_wait_asynccnt(0);
#endif
      __syncthreads();
#endif
      /* rec tile = h_{t-1}[16x512] @ Uᵀ-slice: 16 WMMAs per wave */
      const int arow = (lane & 15) * 520;
#pragma unroll 4
      for (int kb = 0; kb < 16; ++kb) {
        const int k0 = kb * 32;
        v8bf alo = *(const v8bf*)(hbuf + arow + k0 + koff);
        v8bf ahi = *(const v8bf*)(hbuf + arow + k0 + koff + 16);
        v16bf bfr = loadB(ubuf + (k0 + koff) * 136 + ubufCol + myN, 136);
        acc = __builtin_amdgcn_wmma_f32_16x16x32_bf16(
                  false, combine16(alo, ahi), false, bfr,
                  (short)0, acc, false, false);
      }
    }
    __builtin_prefetch(&w[((size_t)rowAdd * T_ + t) * G2H_ + colBase + myN], 0, 1);

    /* LN stats: per-row sum/sumsq over this wave's 16 N columns (C layout:
       lane = N, VGPR r -> M = r + rowAdd) -> shuffle-reduce width 16 */
    float s[8], q[8];
#pragma unroll
    for (int r = 0; r < 8; ++r) { s[r] = acc[r]; q[r] = acc[r] * acc[r]; }
#pragma unroll
    for (int off = 8; off >= 1; off >>= 1) {
#pragma unroll
      for (int r = 0; r < 8; ++r) {
        s[r] += __shfl_xor(s[r], off, 16);
        q[r] += __shfl_xor(q[r], off, 16);
      }
    }
    if (myN == 0) {
#pragma unroll
      for (int r = 0; r < 8; ++r) {
        wpart[wave * 32 + (r + rowAdd) * 2 + 0] = s[r];
        wpart[wave * 32 + (r + rowAdd) * 2 + 1] = q[r];
      }
    }
    __syncthreads();
    const int parity = (int)(((unsigned)layer * T_ + t) & 1u);
    if (tid < 32) {
      float tot = 0.f;
#pragma unroll
      for (int wv = 0; wv < 8; ++wv) tot += wpart[wv * 32 + tid];
      __hip_atomic_store(&partials[(parity * NWG + g) * 32 + tid], tot,
                         __ATOMIC_RELAXED, __HIP_MEMORY_SCOPE_AGENT);
    }
    __threadfence();
    wg_grid_sync(sync1, baseT + (unsigned)(t + 1) * NWG);
    if (tid < 32) {
      float tot = 0.f;
#pragma unroll
      for (int wg2 = 0; wg2 < NWG; ++wg2)
        tot += __hip_atomic_load(&partials[(parity * NWG + wg2) * 32 + tid],
                                 __ATOMIC_RELAXED, __HIP_MEMORY_SCOPE_AGENT);
      stats[tid] = tot;
    }
    __syncthreads();

    /* gates: (rec - mu)*rstd + w  ->  relu / sigmoid halves via LDS */
    const float invN = 1.0f / 1024.0f;
#pragma unroll
    for (int r = 0; r < 8; ++r) {
      const int m = r + rowAdd;
      const float mu   = stats[2 * m] * invN;
      const float var  = stats[2 * m + 1] * invN - mu * mu;
      const float rstd = rsqrtf(var + EPSL);
      const float wval = w[((size_t)m * T_ + t) * G2H_ + colBase + myN];
      const float gate = wval + (acc[r] - mu) * rstd;
      if (isZ) zbuf[m * 64 + lc] = 1.0f / (1.0f + __expf(-gate));
      else     hcand[m * 64 + lc] = fmaxf(gate, 0.0f);
    }
    __syncthreads();

    /* h update for this WG's 16x64 slice; publish bf16 (matmul) + fp32 (output) */
    for (int idx = tid; idx < 16 * 64; idx += 256) {
      const int m = idx >> 6, c = idx & 63;
      const float z  = zbuf[idx];
      const float hn = z * hstate[idx] + (1.0f - z) * hcand[idx];
      hstate[idx] = hn;
      const size_t o = ((size_t)m * T_ + t) * H_ + 64 * g + c;
      hA[o]  = (bf16)hn;
      out[o] = hn;
    }
    __threadfence();
    wg_grid_sync(sync2, baseT + (unsigned)(t + 1) * NWG);
  }
}

/* ---------------- host ---------------- */
extern "C" void kernel_launch(void* const* d_in, const int* in_sizes, int n_in,
                              void* d_out, int out_size, void* d_ws, size_t ws_size,
                              hipStream_t stream) {
  const float* x     = (const float*)d_in[0];
  const float* W     = (const float*)d_in[1];
  const float* U     = (const float*)d_in[2];
  const float* gamma = (const float*)d_in[3];
  const float* beta  = (const float*)d_in[4];
  float* out = (float*)d_out;
  char*  ws  = (char*)d_ws;

  const size_t WMAT_SZ = (size_t)L_ * H_ * G2H_ * sizeof(bf16);  /* 4 MiB */
  const size_t WT_OFF  = 4096;
  const size_t UT_OFF  = WT_OFF + WMAT_SZ;
  const size_t HA_OFF  = UT_OFF + WMAT_SZ;
  const size_t HA_SZ   = (size_t)M_ * H_ * sizeof(bf16);         /* 16.4 MB */
  const size_t WB_OFF  = HA_OFF + HA_SZ;                         /* + 65.5 MB fp32 */

  unsigned* sync1 = (unsigned*)(ws + 0);
  unsigned* sync2 = sync1 + 1;
  float* partials = (float*)(ws + 256);
  bf16*  WT = (bf16*)(ws + WT_OFF);
  bf16*  UT = (bf16*)(ws + UT_OFF);
  bf16*  hA = (bf16*)(ws + HA_OFF);
  float* wb = (float*)(ws + WB_OFF);

  (void)hipMemsetAsync(ws, 0, 4096, stream);  /* sync counters + partials */

  {
    const int n = M_ * H_;
    k_f32_to_bf16<<<(n + 255) / 256, 256, 0, stream>>>(x, hA, n);
    const int nw = L_ * G2H_ * H_;
    k_transpose_to_bf16<<<(nw + 255) / 256, 256, 0, stream>>>(W, WT);
    k_transpose_to_bf16<<<(nw + 255) / 256, 256, 0, stream>>>(U, UT);
  }

  const size_t SMEM = 139264 + 16640 + 3 * 4096 + 1024 + 128;  /* 169,344 B < 320 KB */
  for (int l = 0; l < L_; ++l) {
    k_gemm_ff<<<dim3(125, 8), 256, 0, stream>>>(hA, WT + (size_t)l * H_ * G2H_, wb);
    k_ln_rows<<<M_, 256, 0, stream>>>(wb, gamma + l * G2H_, beta + l * G2H_);
    k_recur<<<NWG, 256, SMEM, stream>>>(UT + (size_t)l * H_ * G2H_, wb, hA, out,
                                        partials, sync1, sync2, l);
  }
  (void)in_sizes; (void)n_in; (void)out_size; (void)ws_size;
}
